// CenterShift_12833362280550
// MI455X (gfx1250) — matrix operand order repaired and checked by reading, not compiled, not run on hardware
//
#include <hip/hip_runtime.h>

typedef __attribute__((ext_vector_type(16))) _Float16 v16h;
typedef __attribute__((ext_vector_type(8)))  float    v8f;

#define H2_STRIDE 68   // 64 cols + 4 pad floats: conflict-free banks, 16B-aligned rows

__device__ __forceinline__ float celu1(float t) {
    // jax.nn.celu, alpha=1:  t>0 ? t : expm1(t)
    return t > 0.0f ? t : (__expf(t) - 1.0f);
}

__global__ __launch_bounds__(256, 1)
void pointconv_dynw_kernel(const float* __restrict__ x,
                           const float* __restrict__ pos_i,
                           const float* __restrict__ pos_j,
                           const float* __restrict__ w11, const float* __restrict__ b11,
                           const float* __restrict__ w12, const float* __restrict__ b12,
                           const float* __restrict__ w13, const float* __restrict__ b13,
                           float* __restrict__ out, int n)
{
    const int lane   = threadIdx.x & 31;
    const int waveIn = threadIdx.x >> 5;
    const int col    = lane & 15;   // matrix column / point-in-tile
    const int hi     = lane >> 4;   // which half of the wave

    // per-wave staging tile: h2 [16 rows x 64 cols], padded
    __shared__ float lds_h2[8][16 * H2_STRIDE];
    float* h2s = lds_h2[waveIn];

    // ------- hoist weights into per-lane WMMA layouts (once per wave) -------
    // layer1: this lane produces h1[k], k = hi*8 + j, for point `col`
    float w11c[8][3], b11c[8];
#pragma unroll
    for (int j = 0; j < 8; ++j) {
        const int k = hi * 8 + j;
        w11c[j][0] = w11[0 * 16 + k];
        w11c[j][1] = w11[1 * 16 + k];
        w11c[j][2] = w11[2 * 16 + k];
        b11c[j]    = b11[k];
    }

    // layer2 B operand: w12 [16 x 64]; element e<8 -> K=hi*8+e, elements 8..15 (K>=16) zero
    v16h Bw12[4];
#pragma unroll
    for (int c = 0; c < 4; ++c) {
        v16h b = {};
        {
#pragma unroll
            for (int e = 0; e < 8; ++e)
                b[e] = (_Float16)w12[(hi * 8 + e) * 64 + c * 16 + col];
        }
        Bw12[c] = b;
    }
    float b12c[4], b13c[4];
#pragma unroll
    for (int c = 0; c < 4; ++c) { b12c[c] = b12[c * 16 + col]; b13c[c] = b13[c * 16 + col]; }

    // layer3 B operand: w13 [64 x 64]; B 32x16 layout: element e -> K=kc*32+(e/8)*16+hi*8+(e%8)
    v16h Bw13[2][4];
#pragma unroll
    for (int kc = 0; kc < 2; ++kc)
#pragma unroll
        for (int c = 0; c < 4; ++c) {
            v16h b;
#pragma unroll
            for (int e = 0; e < 16; ++e) {
                const int K = kc * 32 + (e >> 3) * 16 + hi * 8 + (e & 7);
                b[e] = (_Float16)w13[K * 64 + c * 16 + col];
            }
            Bw13[kc][c] = b;
        }

    // ------- grid-stride over 16-point tiles (loop condition is wave-uniform) -------
    const int nwaves = gridDim.x * (blockDim.x >> 5);
    const int wave   = blockIdx.x * (blockDim.x >> 5) + waveIn;
    const int ntiles = n >> 4;

    for (int tile = wave; tile < ntiles; tile += nwaves) {
        const int base = tile << 4;
        const int p    = base + col;

        // prefetch next tile's inputs (global_prefetch_b8)
        const int nb = base + nwaves * 16;
        if (nb < n) {
            __builtin_prefetch(&pos_i[(nb + col) * 3], 0, 1);
            __builtin_prefetch(&pos_j[(nb + col) * 3], 0, 1);
            __builtin_prefetch(&x[nb + col], 0, 1);
        }

        // ---- layer 1 (scalar VALU), result lands in A layout for 16x16x32 f16 ----
        const float px0 = pos_j[p * 3 + 0] - pos_i[p * 3 + 0];
        const float px1 = pos_j[p * 3 + 1] - pos_i[p * 3 + 1];
        const float px2 = pos_j[p * 3 + 2] - pos_i[p * 3 + 2];
        v16h A = {};   // elements 8..15 stay zero (K=16..31 padding)
#pragma unroll
        for (int j = 0; j < 8; ++j) {
            float t = fmaf(px0, w11c[j][0],
                      fmaf(px1, w11c[j][1],
                      fmaf(px2, w11c[j][2], b11c[j])));
            A[j] = (_Float16)celu1(t);
        }

        // ---- layer 2: h2 = celu(h1 @ w12 + b12), 4 WMMAs ----
        v8f h2acc[4];
#pragma unroll
        for (int c = 0; c < 4; ++c) {
            v8f cc;
#pragma unroll
            for (int v = 0; v < 8; ++v) cc[v] = b12c[c];
            h2acc[c] = __builtin_amdgcn_wmma_f32_16x16x32_f16(
                false, A, false, Bw12[c], (short)0, cc, false, false);
        }

        // stage celu(h2) to LDS in row-major [row][col] (C/D layout -> A layout transpose)
#pragma unroll
        for (int c = 0; c < 4; ++c)
#pragma unroll
            for (int v = 0; v < 8; ++v)
                h2s[(v + 8 * hi) * H2_STRIDE + c * 16 + col] = celu1(h2acc[c][v]);

        // rebuild A operands for layer 3 (row m = col), two K-chunks of 32
        v16h A3[2];
#pragma unroll
        for (int kc = 0; kc < 2; ++kc) {
            const float* rowp = &h2s[col * H2_STRIDE + kc * 32 + hi * 8];
            v16h a;
#pragma unroll
            for (int e = 0; e < 8; ++e) {
                a[e]     = (_Float16)rowp[e];        // K = kc*32 + hi*8 + e
                a[e + 8] = (_Float16)rowp[16 + e];   // K = kc*32 + 16 + hi*8 + e
            }
            A3[kc] = a;
        }

        // ---- layer 3: W = h2 @ w13 + b13, 8 chained WMMAs, then out = x * W ----
#pragma unroll
        for (int c = 0; c < 4; ++c) {
            v8f cc;
#pragma unroll
            for (int v = 0; v < 8; ++v) cc[v] = b13c[c];
            cc = __builtin_amdgcn_wmma_f32_16x16x32_f16(
                false, A3[0], false, Bw13[0][c], (short)0, cc, false, false);
            cc = __builtin_amdgcn_wmma_f32_16x16x32_f16(
                false, A3[1], false, Bw13[1][c], (short)0, cc, false, false);
#pragma unroll
            for (int v = 0; v < 8; ++v) {
                const int m = v + 8 * hi;                       // row in tile
                out[(size_t)(base + m) * 64 + c * 16 + col] = x[base + m] * cc[v];
            }
        }
    }
}

extern "C" void kernel_launch(void* const* d_in, const int* in_sizes, int n_in,
                              void* d_out, int out_size, void* d_ws, size_t ws_size,
                              hipStream_t stream) {
    const float* x     = (const float*)d_in[0];
    const float* pos_i = (const float*)d_in[1];
    const float* pos_j = (const float*)d_in[2];
    const float* w11   = (const float*)d_in[3];
    const float* b11   = (const float*)d_in[4];
    const float* w12   = (const float*)d_in[5];
    const float* b12   = (const float*)d_in[6];
    const float* w13   = (const float*)d_in[7];
    const float* b13   = (const float*)d_in[8];
    float* out = (float*)d_out;

    const int n = in_sizes[0];   // N (C_IN == 1)

    dim3 block(256);             // 8 waves per block (wave32)
    dim3 grid(1024);             // 8192 waves, ~16 tiles each (grid-stride)
    pointconv_dynw_kernel<<<grid, block, 0, stream>>>(
        x, pos_i, pos_j, w11, b11, w12, b12, w13, b13, out, n);
}
